// SpatialMHA_44109314130349
// MI455X (gfx1250) — matrix operand order repaired and checked by reading, not compiled, not run on hardware
//
#include <hip/hip_runtime.h>

// Spatial MHA forward for MI455X (gfx1250, wave32, WMMA, async-to-LDS).
// B=64, S=256, E=1024, H=16, D=64.
// Workspace layout (bytes), ~176 MB total:
//   [0          ) x_bf16     [16384,1024]        33,554,432
//   [33554432   ) w_in_bf16  [3072,1024]          6,291,456
//   [39845888   ) w_out_bf16 [1024,1024]          2,097,152
//   [41943040   ) q_bf16     [B,H,S,D] (scaled)  33,554,432
//   [75497472   ) k_bf16     [B,H,S,D]           33,554,432
//   [109051904  ) v_bf16     [B,H,S,D]           33,554,432
//   [142606336  ) ctx_bf16   [B,S,E]             33,554,432

#define B_ 64
#define S_ 256
#define E_ 1024
#define H_ 16
#define D_ 64

typedef __attribute__((ext_vector_type(16))) __bf16 bf16x16;
typedef __attribute__((ext_vector_type(8)))  float  f32x8;

static __device__ __forceinline__ f32x8 wmma_bf16(bf16x16 a, bf16x16 b, f32x8 c) {
  // D = A(16x32 bf16) * B(32x16 bf16) + C(16x16 f32)
  return __builtin_amdgcn_wmma_f32_16x16x32_bf16(false, a, false, b, (short)0, c,
                                                 false, false);
}

// Low 32 bits of a generic pointer into __shared__ are the LDS byte offset
// (ISA 10.2: LDS aperture address is truncated to addr[31:0]).
static __device__ __forceinline__ unsigned lds_off(const void* p) {
  return (unsigned)(unsigned long long)p;
}

// CDNA5 async direct-to-LDS copy, 16B per lane, tracked with ASYNCcnt.
static __device__ __forceinline__ void async_ld_b128(unsigned lds, const void* g) {
  asm volatile("global_load_async_to_lds_b128 %0, %1, off"
               :
               : "v"(lds), "v"((unsigned long long)g)
               : "memory");
}

static __device__ __forceinline__ void wait_async0() {
  asm volatile("s_wait_asynccnt 0x0" ::: "memory");
}

// ---------------------------------------------------------------- fp32 -> bf16
__global__ void cvt_f32_bf16(const float* __restrict__ in, __bf16* __restrict__ out,
                             int n) {
  for (int i = blockIdx.x * blockDim.x + threadIdx.x; i < n;
       i += gridDim.x * blockDim.x)
    out[i] = (__bf16)in[i];
}

// ------------------------------------------------------------------- QKV GEMM
// C[16384,3072] = Xbf16[16384,1024] * Win^T (Win [3072,1024] row-major).
// Block tile 128x128, 8 waves, each wave 32(M) x 64(N), BK=32.
// Double-buffered LDS fed by async-to-LDS loads: one barrier per K step.
__global__ __launch_bounds__(256)
void qkv_gemm(const __bf16* __restrict__ A, const __bf16* __restrict__ W,
              const float* __restrict__ bias, __bf16* __restrict__ q,
              __bf16* __restrict__ k, __bf16* __restrict__ v) {
  __shared__ __bf16 As[2][128][40];  // +8 pad: row stride 80B, conflict-free frags
  __shared__ __bf16 Bs[2][128][40];
  const int K = 1024;
  const int t = threadIdx.x, lane = t & 31, w = t >> 5;
  const int wm = w >> 1, wn = w & 1;
  const int m0 = blockIdx.y * 128, n0 = blockIdx.x * 128;
  const int lrow = t >> 1, lch = (t & 1) * 16;
  const __bf16* ga = A + (size_t)(m0 + lrow) * K + lch;
  const __bf16* gw = W + (size_t)(n0 + lrow) * K + lch;

  const f32x8 zero = {0.f, 0.f, 0.f, 0.f, 0.f, 0.f, 0.f, 0.f};
  f32x8 acc[2][4];
#pragma unroll
  for (int i = 0; i < 2; ++i)
#pragma unroll
    for (int j = 0; j < 4; ++j) acc[i][j] = zero;

  const int lm = lane & 15;
  const int kA = (lane >> 4) * 8;   // A frag: k in {kA..kA+7} and {kA+16..kA+23}
  const int kB = (lane >> 4) * 8;   // B frag: k in {kB..kB+7} and {kB+16..kB+23}

  // Prologue: async-stage first K tile into buffer 0.
  async_ld_b128(lds_off(&As[0][lrow][lch]),     ga);
  async_ld_b128(lds_off(&As[0][lrow][lch + 8]), ga + 8);
  async_ld_b128(lds_off(&Bs[0][lrow][lch]),     gw);
  async_ld_b128(lds_off(&Bs[0][lrow][lch + 8]), gw + 8);

  int cur = 0;
  for (int kk = 0; kk < K; kk += 32, cur ^= 1) {
    wait_async0();       // current buffer's async copies landed in LDS
    __syncthreads();     // all waves done reading the buffer we overwrite next
    if (kk + 32 < K) {   // kick next tile into the other buffer (overlaps WMMA)
      async_ld_b128(lds_off(&As[cur ^ 1][lrow][lch]),     ga + kk + 32);
      async_ld_b128(lds_off(&As[cur ^ 1][lrow][lch + 8]), ga + kk + 40);
      async_ld_b128(lds_off(&Bs[cur ^ 1][lrow][lch]),     gw + kk + 32);
      async_ld_b128(lds_off(&Bs[cur ^ 1][lrow][lch + 8]), gw + kk + 40);
      __builtin_prefetch(ga + kk + 64, 0, 1);  // gfx1250 global_prefetch_b8
      __builtin_prefetch(gw + kk + 64, 0, 1);
    }
    bf16x16 af[2], bw[4];
#pragma unroll
    for (int mt = 0; mt < 2; ++mt) {
      const int r = wm * 32 + mt * 16 + lm;
      *(uint4*)&af[mt]       = *(const uint4*)&As[cur][r][kA];
      *((uint4*)&af[mt] + 1) = *(const uint4*)&As[cur][r][kA + 16];
    }
#pragma unroll
    for (int nt = 0; nt < 4; ++nt) {
      const int c = wn * 64 + nt * 16 + lm;
      *(uint4*)&bw[nt]       = *(const uint4*)&Bs[cur][c][kB];
      *((uint4*)&bw[nt] + 1) = *(const uint4*)&Bs[cur][c][kB + 16];
    }
#pragma unroll
    for (int mt = 0; mt < 2; ++mt)
#pragma unroll
      for (int nt = 0; nt < 4; ++nt)
        acc[mt][nt] = wmma_bf16(af[mt], bw[nt], acc[mt][nt]);
  }

  // Epilogue: +bias, fold 1/sqrt(D)=0.125 into q, scatter to [B,H,S,D] bf16.
  const int lh = (lane >> 4) << 3;  // C tile: lanes>=16 hold rows M=8..15
#pragma unroll
  for (int mt = 0; mt < 2; ++mt) {
    const int rbase = m0 + wm * 32 + mt * 16 + lh;
#pragma unroll
    for (int nt = 0; nt < 4; ++nt) {
      const int n = n0 + wn * 64 + nt * 16 + lm;
      const int seg = n >> 10, e = n & 1023, hh = e >> 6, dd = e & 63;
      __bf16* dst = (seg == 0) ? q : (seg == 1) ? k : v;
      const float bn = bias[n];
      const float sc = (seg == 0) ? 0.125f : 1.0f;
#pragma unroll
      for (int rr = 0; rr < 8; ++rr) {
        const int row = rbase + rr;
        const int bb = row >> 8, ss = row & 255;
        dst[(((size_t)bb * H_ + hh) * S_ + ss) * D_ + dd] =
            (__bf16)((acc[mt][nt][rr] + bn) * sc);
      }
    }
  }
}

// -------------------------------------------------------------- fused attention
// One block = one (b,h) pair x 64 query rows; 4 waves, 16 query rows per wave.
// Full 16x256 score strip kept in VGPRs -> exact softmax, no score tensor in HBM.
__global__ __launch_bounds__(128)
void attn_fused(const __bf16* __restrict__ Q, const __bf16* __restrict__ Km,
                const __bf16* __restrict__ V, const float* __restrict__ bias,
                __bf16* __restrict__ ctx) {
  __shared__ __bf16 VtS[D_ * S_];       // V transposed [d][s], 32 KB
  __shared__ __bf16 PsS[4 * 16 * S_];   // per-wave P rows, bf16, 32 KB
  const int t = threadIdx.x, lane = t & 31, w = t >> 5;
  const int bh = blockIdx.x >> 2, qblk = blockIdx.x & 3;
  const int b = bh >> 4, h = bh & 15;
  const __bf16* Qg = Q + (size_t)bh * S_ * D_;
  const __bf16* Kg = Km + (size_t)bh * S_ * D_;
  const __bf16* Vg = V + (size_t)bh * S_ * D_;

  // Stage V transposed into LDS (enables contiguous ds_load_b128 B-fragments).
  for (int i = t; i < S_ * D_; i += 128) {
    const int s = i >> 6, d = i & 63;
    VtS[d * S_ + s] = Vg[i];
  }
  __syncthreads();

  const int lm = lane & 15;
  const int kA = (lane >> 4) * 8;
  const int kB = (lane >> 4) * 8;
  const int lh = (lane >> 4) << 3;
  const int qrow = qblk * 64 + w * 16;

  // Q fragments for the two d-chunks (0..31, 32..63); q pre-scaled by 0.125.
  bf16x16 a0, a1;
  const __bf16* qp = Qg + (size_t)(qrow + lm) * D_;
  *(uint4*)&a0       = *(const uint4*)(qp + kA);
  *((uint4*)&a0 + 1) = *(const uint4*)(qp + kA + 16);
  *(uint4*)&a1       = *(const uint4*)(qp + 32 + kA);
  *((uint4*)&a1 + 1) = *(const uint4*)(qp + 32 + kA + 16);

  const f32x8 zero = {0.f, 0.f, 0.f, 0.f, 0.f, 0.f, 0.f, 0.f};

  // scores = Q K^T (scaled) + spatial_bias, 16 key tiles of 16.
  f32x8 sc[16];
#pragma unroll
  for (int kt = 0; kt < 16; ++kt) {
    f32x8 acc = zero;
    const __bf16* kp = Kg + (size_t)(kt * 16 + lm) * D_;
    bf16x16 bk;
    *(uint4*)&bk       = *(const uint4*)(kp + kB);
    *((uint4*)&bk + 1) = *(const uint4*)(kp + kB + 16);
    acc = wmma_bf16(a0, bk, acc);
    *(uint4*)&bk       = *(const uint4*)(kp + 32 + kB);
    *((uint4*)&bk + 1) = *(const uint4*)(kp + 32 + kB + 16);
    acc = wmma_bf16(a1, bk, acc);
    const int col = kt * 16 + lm;
#pragma unroll
    for (int rr = 0; rr < 8; ++rr)
      acc[rr] += bias[(size_t)(qrow + lh + rr) * S_ + col];
    sc[kt] = acc;
  }

  // Exact row softmax: reduce over 16 tiles, then across the 16-lane row group.
  float mx[8], sum[8], rinv[8];
#pragma unroll
  for (int rr = 0; rr < 8; ++rr) mx[rr] = -3.0e38f;
#pragma unroll
  for (int kt = 0; kt < 16; ++kt)
#pragma unroll
    for (int rr = 0; rr < 8; ++rr) mx[rr] = fmaxf(mx[rr], sc[kt][rr]);
#pragma unroll
  for (int rr = 0; rr < 8; ++rr)
    for (int off = 1; off < 16; off <<= 1)
      mx[rr] = fmaxf(mx[rr], __shfl_xor(mx[rr], off, 32));
#pragma unroll
  for (int rr = 0; rr < 8; ++rr) sum[rr] = 0.f;
#pragma unroll
  for (int kt = 0; kt < 16; ++kt)
#pragma unroll
    for (int rr = 0; rr < 8; ++rr) {
      const float e = __expf(sc[kt][rr] - mx[rr]);
      sc[kt][rr] = e;
      sum[rr] += e;
    }
#pragma unroll
  for (int rr = 0; rr < 8; ++rr)
    for (int off = 1; off < 16; off <<= 1)
      sum[rr] += __shfl_xor(sum[rr], off, 32);
#pragma unroll
  for (int rr = 0; rr < 8; ++rr) rinv[rr] = 1.0f / sum[rr];

  // Normalize and spill P (bf16) to this wave's LDS strip.
#pragma unroll
  for (int kt = 0; kt < 16; ++kt)
#pragma unroll
    for (int rr = 0; rr < 8; ++rr)
      PsS[(w * 16 + lh + rr) * S_ + kt * 16 + lm] =
          (__bf16)(sc[kt][rr] * rinv[rr]);
  __syncthreads();

  // ctx(16x64) = P(16x256) * V(256x64)
  f32x8 cacc[4];
#pragma unroll
  for (int dt = 0; dt < 4; ++dt) cacc[dt] = zero;
#pragma unroll
  for (int kc = 0; kc < 8; ++kc) {
    bf16x16 pa;
    const __bf16* pp = &PsS[(w * 16 + lm) * S_ + kc * 32];
    *(uint4*)&pa       = *(const uint4*)(pp + kA);
    *((uint4*)&pa + 1) = *(const uint4*)(pp + kA + 16);
#pragma unroll
    for (int dt = 0; dt < 4; ++dt) {
      bf16x16 vb;
      const __bf16* vp = &VtS[(dt * 16 + lm) * S_ + kc * 32];
      *(uint4*)&vb       = *(const uint4*)(vp + kB);
      *((uint4*)&vb + 1) = *(const uint4*)(vp + kB + 16);
      cacc[dt] = wmma_bf16(pa, vb, cacc[dt]);
    }
  }

  // Store ctx as [B,S,E] bf16 (feeds the out-proj GEMM as its A matrix).
#pragma unroll
  for (int dt = 0; dt < 4; ++dt)
#pragma unroll
    for (int rr = 0; rr < 8; ++rr) {
      const int s = qrow + lh + rr;
      const int d = dt * 16 + lm;
      ctx[((size_t)b * S_ + s) * E_ + h * D_ + d] = (__bf16)cacc[dt][rr];
    }
}

// --------------------------------------------------- out-proj GEMM + residual
__global__ __launch_bounds__(256)
void out_gemm(const __bf16* __restrict__ A, const __bf16* __restrict__ W,
              const float* __restrict__ bias, const float* __restrict__ xres,
              float* __restrict__ out) {
  __shared__ __bf16 As[2][128][40];
  __shared__ __bf16 Bs[2][128][40];
  const int K = 1024;
  const int t = threadIdx.x, lane = t & 31, w = t >> 5;
  const int wm = w >> 1, wn = w & 1;
  const int m0 = blockIdx.y * 128, n0 = blockIdx.x * 128;
  const int lrow = t >> 1, lch = (t & 1) * 16;
  const __bf16* ga = A + (size_t)(m0 + lrow) * K + lch;
  const __bf16* gw = W + (size_t)(n0 + lrow) * K + lch;

  const f32x8 zero = {0.f, 0.f, 0.f, 0.f, 0.f, 0.f, 0.f, 0.f};
  f32x8 acc[2][4];
#pragma unroll
  for (int i = 0; i < 2; ++i)
#pragma unroll
    for (int j = 0; j < 4; ++j) acc[i][j] = zero;

  const int lm = lane & 15;
  const int kA = (lane >> 4) * 8;
  const int kB = (lane >> 4) * 8;

  async_ld_b128(lds_off(&As[0][lrow][lch]),     ga);
  async_ld_b128(lds_off(&As[0][lrow][lch + 8]), ga + 8);
  async_ld_b128(lds_off(&Bs[0][lrow][lch]),     gw);
  async_ld_b128(lds_off(&Bs[0][lrow][lch + 8]), gw + 8);

  int cur = 0;
  for (int kk = 0; kk < K; kk += 32, cur ^= 1) {
    wait_async0();
    __syncthreads();
    if (kk + 32 < K) {
      async_ld_b128(lds_off(&As[cur ^ 1][lrow][lch]),     ga + kk + 32);
      async_ld_b128(lds_off(&As[cur ^ 1][lrow][lch + 8]), ga + kk + 40);
      async_ld_b128(lds_off(&Bs[cur ^ 1][lrow][lch]),     gw + kk + 32);
      async_ld_b128(lds_off(&Bs[cur ^ 1][lrow][lch + 8]), gw + kk + 40);
      __builtin_prefetch(ga + kk + 64, 0, 1);
      __builtin_prefetch(gw + kk + 64, 0, 1);
    }
    bf16x16 af[2], bw[4];
#pragma unroll
    for (int mt = 0; mt < 2; ++mt) {
      const int r = wm * 32 + mt * 16 + lm;
      *(uint4*)&af[mt]       = *(const uint4*)&As[cur][r][kA];
      *((uint4*)&af[mt] + 1) = *(const uint4*)&As[cur][r][kA + 16];
    }
#pragma unroll
    for (int nt = 0; nt < 4; ++nt) {
      const int c = wn * 64 + nt * 16 + lm;
      *(uint4*)&bw[nt]       = *(const uint4*)&Bs[cur][c][kB];
      *((uint4*)&bw[nt] + 1) = *(const uint4*)&Bs[cur][c][kB + 16];
    }
#pragma unroll
    for (int mt = 0; mt < 2; ++mt)
#pragma unroll
      for (int nt = 0; nt < 4; ++nt)
        acc[mt][nt] = wmma_bf16(af[mt], bw[nt], acc[mt][nt]);
  }

  const int lh = (lane >> 4) << 3;
#pragma unroll
  for (int mt = 0; mt < 2; ++mt) {
    const int rbase = m0 + wm * 32 + mt * 16 + lh;
#pragma unroll
    for (int nt = 0; nt < 4; ++nt) {
      const int n = n0 + wn * 64 + nt * 16 + lm;
      const float bn = bias[n];
#pragma unroll
      for (int rr = 0; rr < 8; ++rr) {
        const int row = rbase + rr;
        out[(size_t)row * E_ + n] =
            acc[mt][nt][rr] + bn + xres[(size_t)row * E_ + n];
      }
    }
  }
}

// -------------------------------------------------------------------- launcher
extern "C" void kernel_launch(void* const* d_in, const int* in_sizes, int n_in,
                              void* d_out, int out_size, void* d_ws, size_t ws_size,
                              hipStream_t stream) {
  const float* x  = (const float*)d_in[0];
  const float* wi = (const float*)d_in[1];
  const float* bi = (const float*)d_in[2];
  const float* wo = (const float*)d_in[3];
  const float* bo = (const float*)d_in[4];
  const float* sb = (const float*)d_in[5];
  float* out = (float*)d_out;

  char* ws = (char*)d_ws;
  __bf16* xb  = (__bf16*)(ws);
  __bf16* wib = (__bf16*)(ws + 33554432);
  __bf16* wob = (__bf16*)(ws + 39845888);
  __bf16* qb  = (__bf16*)(ws + 41943040);
  __bf16* kb  = (__bf16*)(ws + 75497472);
  __bf16* vb  = (__bf16*)(ws + 109051904);
  __bf16* cb  = (__bf16*)(ws + 142606336);

  cvt_f32_bf16<<<2048, 256, 0, stream>>>(x, xb, B_ * S_ * E_);
  cvt_f32_bf16<<<512, 256, 0, stream>>>(wi, wib, 3 * E_ * E_);
  cvt_f32_bf16<<<256, 256, 0, stream>>>(wo, wob, E_ * E_);

  qkv_gemm<<<dim3(24, 128), 256, 0, stream>>>(xb, wib, bi, qb, kb, vb);
  attn_fused<<<dim3(B_ * H_ * 4), 128, 0, stream>>>(qb, kb, vb, sb, cb);
  out_gemm<<<dim3(8, 128), 256, 0, stream>>>(cb, wob, bo, x, out);
}
